// MLP_16234976379523
// MI455X (gfx1250) — compile-verified, matrix operand order checked by reading, not run on hardware
//
#include <hip/hip_runtime.h>
#include <hip/hip_bf16.h>
#include <math.h>

// CDNA5 wave32 WMMA types
typedef __attribute__((ext_vector_type(16))) __bf16 v16bf;
typedef __attribute__((ext_vector_type(8)))  float  v8f;

#define N_NODES 100000
#define IN_DIM  256
#define HID     128
#define OUT_DIM 64
#define EPB     16   // edges per block in spmm

// ---------------------------------------------------------------------------
// Pack a fp32 row-major weight [K, Ncols] into bf16 WMMA-B register order:
// element index = ((tile*chunks + kc)*32 + lane)*16 + i, where
//   tile = N-tile (16 cols), kc = K-chunk (32), lane = wave lane,
//   n = lane&15, kbase = (lane>>4)*8,
//   i<8  -> K = kc*32 + kbase + i
//   i>=8 -> K = kc*32 + kbase + 16 + (i-8)
// (mirrors ISA 7.12.2 16-bit A/B layout). A wave then fetches its whole
// 32x16 B chunk as one contiguous 32-byte load (2x global_load_b128).
__global__ __launch_bounds__(256) void pack_wmma_b(const float* __restrict__ W,
                                                   __bf16* __restrict__ Wp,
                                                   int K, int Ncols) {
    int idx = blockIdx.x * 256 + threadIdx.x;
    if (idx >= K * Ncols) return;
    const int chunks = K >> 5;
    const int i     = idx & 15;
    const int lane  = (idx >> 4) & 31;
    const int kc    = (idx >> 9) % chunks;
    const int t     = idx / (512 * chunks);
    const int n     = lane & 15;
    const int kbase = (lane >> 4) * 8;
    const int k     = kc * 32 + kbase + (i < 8 ? i : 16 + (i - 8));
    Wp[idx] = (__bf16)W[(size_t)k * Ncols + t * 16 + n];
}

// ---------------------------------------------------------------------------
// fc1: H1b[N,128] = bf16( X[N,256] @ W1 + b1 )
// Block = 256 threads = 8 waves; wave w owns the 16x16 tile (rows blockIdx*16,
// cols w*16). K swept in 8 chunks of 32 via v_wmma_f32_16x16x32_bf16.
// A loaded from fp32 features (2x b128 per 8-float group, cvt_pk to bf16);
// B loaded pre-packed (2x b128 per chunk).
__global__ __launch_bounds__(256) void fc1_wmma(const float*  __restrict__ X,
                                                const __bf16* __restrict__ W1p,
                                                const float*  __restrict__ b1,
                                                __bf16*       __restrict__ H1b) {
    const int lane  = threadIdx.x & 31;
    const int wave  = threadIdx.x >> 5;
    const int row0  = blockIdx.x * 16;
    const int col0  = wave * 16;
    const int m     = lane & 15;     // M (for A) / N (for B/C)
    const int half  = lane >> 4;     // 0: K-base 0, 1: K-base 8
    const int kbase = half * 8;

    v8f c = {};
    const float* arow = X + (size_t)(row0 + m) * IN_DIM + kbase;
    const v16bf* bp   = (const v16bf*)W1p + (size_t)wave * (IN_DIM / 32) * 32 + lane;

    #pragma unroll
    for (int kc = 0; kc < IN_DIM / 32; ++kc) {
        const int k0 = kc * 32;
        v16bf a;
        #pragma unroll
        for (int i = 0; i < 8; ++i) {
            a[i]     = (__bf16)arow[k0 + i];
            a[i + 8] = (__bf16)arow[k0 + 16 + i];
        }
        const v16bf b = bp[(size_t)kc * 32];
        c = __builtin_amdgcn_wmma_f32_16x16x32_bf16(false, a, false, b,
                                                    (short)0, c, false, false);
    }

    // C/D layout: lane<16 -> (M=r, N=lane); lane>=16 -> (M=8+r, N=lane-16)
    const float bias = b1[col0 + m];
    __bf16* out = H1b + (size_t)(row0 + half * 8) * HID + col0 + m;
    #pragma unroll
    for (int r = 0; r < 8; ++r)
        out[(size_t)r * HID] = (__bf16)(c[r] + bias);
}

// ---------------------------------------------------------------------------
__global__ __launch_bounds__(256) void zero_f32(float* __restrict__ p, int n) {
    int i = blockIdx.x * 256 + threadIdx.x;
    if (i < n) p[i] = 0.0f;
}

// ---------------------------------------------------------------------------
// SpMM scatter: for each edge (r,c,v): H2[r,:] += v * H1b[c,:]
// 128 threads = one feature vector (coalesced); EPB edges per block.
// H1b (25.6 MB) is L2-resident (192 MB L2), so the random gather hits L2.
// The scatter-add uses the CDNA5 hardware FP32 atomic via inline asm
// (GLOBAL_ATOMIC_ADD_F32, non-returning -> STOREcnt; S_ENDPGM's implicit
// wait-idle covers completion).
__global__ __launch_bounds__(128) void spmm_scatter(const int*    __restrict__ erow,
                                                    const int*    __restrict__ ecol,
                                                    const float*  __restrict__ eval,
                                                    const __bf16* __restrict__ H1b,
                                                    float*        __restrict__ H2,
                                                    int E) {
    const int d     = threadIdx.x;
    const int ebase = blockIdx.x * EPB;
    #pragma unroll 1
    for (int j = 0; j < EPB; ++j) {
        const int e = ebase + j;
        if (e >= E) break;
        const int   r  = erow[e];
        const int   cc = ecol[e];
        const float v  = eval[e];
        const float h  = (float)H1b[(size_t)cc * HID + d];
        float* p = &H2[(size_t)r * HID + d];
        const float upd = v * h;
        asm volatile("global_atomic_add_f32 %0, %1, off"
                     :
                     : "v"(p), "v"(upd)
                     : "memory");
    }
}

// ---------------------------------------------------------------------------
// fc2 + log_softmax: Out[N,64] = log_softmax( relu(H2) @ W2 + b2 )
// Block = 128 threads = 4 waves; wave w owns cols w*16..w*16+15 of one 16-row
// tile. ReLU fused into the A-load. Logits staged in LDS; 16 threads then do
// the 64-wide log-softmax per row.
__global__ __launch_bounds__(128) void fc2_logsoftmax(const float*  __restrict__ H2,
                                                      const __bf16* __restrict__ W2p,
                                                      const float*  __restrict__ b2,
                                                      float*        __restrict__ Out) {
    __shared__ float lds[16][OUT_DIM + 4];

    const int lane  = threadIdx.x & 31;
    const int wave  = threadIdx.x >> 5;
    const int row0  = blockIdx.x * 16;
    const int col0  = wave * 16;
    const int m     = lane & 15;
    const int half  = lane >> 4;
    const int kbase = half * 8;

    v8f c = {};
    const float* arow = H2 + (size_t)(row0 + m) * HID + kbase;
    const v16bf* bp   = (const v16bf*)W2p + (size_t)wave * (HID / 32) * 32 + lane;

    #pragma unroll
    for (int kc = 0; kc < HID / 32; ++kc) {
        const int k0 = kc * 32;
        v16bf a;
        #pragma unroll
        for (int i = 0; i < 8; ++i) {
            const float x0 = arow[k0 + i];
            const float x1 = arow[k0 + 16 + i];
            a[i]     = (__bf16)(x0 > 0.0f ? x0 : 0.0f);
            a[i + 8] = (__bf16)(x1 > 0.0f ? x1 : 0.0f);
        }
        const v16bf b = bp[(size_t)kc * 32];
        c = __builtin_amdgcn_wmma_f32_16x16x32_bf16(false, a, false, b,
                                                    (short)0, c, false, false);
    }

    const float bias = b2[col0 + m];
    #pragma unroll
    for (int r = 0; r < 8; ++r)
        lds[half * 8 + r][col0 + m] = c[r] + bias;

    __syncthreads();

    if (threadIdx.x < 16) {
        const int row = threadIdx.x;
        float mx = -INFINITY;
        #pragma unroll
        for (int dd = 0; dd < OUT_DIM; ++dd) mx = fmaxf(mx, lds[row][dd]);
        float s = 0.0f;
        #pragma unroll
        for (int dd = 0; dd < OUT_DIM; ++dd) s += __expf(lds[row][dd] - mx);
        const float lse = mx + __logf(s);
        float* orow = Out + (size_t)(row0 + row) * OUT_DIM;
        #pragma unroll
        for (int dd = 0; dd < OUT_DIM; ++dd) orow[dd] = lds[row][dd] - lse;
    }
}

// ---------------------------------------------------------------------------
extern "C" void kernel_launch(void* const* d_in, const int* in_sizes, int n_in,
                              void* d_out, int out_size, void* d_ws, size_t ws_size,
                              hipStream_t stream) {
    const float* features = (const float*)d_in[0];
    const int*   erow     = (const int*)  d_in[1];
    const int*   ecol     = (const int*)  d_in[2];
    const float* eval     = (const float*)d_in[3];
    const float* W1       = (const float*)d_in[4];
    const float* b1       = (const float*)d_in[5];
    const float* W2       = (const float*)d_in[6];
    const float* b2       = (const float*)d_in[7];
    float*       Out      = (float*)d_out;
    const int    E        = in_sizes[1];

    // Workspace carve-up (~77 MB total)
    char* ws = (char*)d_ws;
    size_t off = 0;
    __bf16* H1b = (__bf16*)(ws + off);
    off += (size_t)N_NODES * HID * sizeof(__bf16); off = (off + 255) & ~(size_t)255;
    float*  H2  = (float*)(ws + off);
    off += (size_t)N_NODES * HID * sizeof(float);  off = (off + 255) & ~(size_t)255;
    __bf16* W1p = (__bf16*)(ws + off);
    off += (size_t)IN_DIM * HID * sizeof(__bf16);  off = (off + 255) & ~(size_t)255;
    __bf16* W2p = (__bf16*)(ws + off);

    pack_wmma_b<<<(IN_DIM * HID + 255) / 256, 256, 0, stream>>>(W1, W1p, IN_DIM, HID);
    pack_wmma_b<<<(HID * OUT_DIM + 255) / 256, 256, 0, stream>>>(W2, W2p, HID, OUT_DIM);
    fc1_wmma<<<N_NODES / 16, 256, 0, stream>>>(features, W1p, b1, H1b);
    zero_f32<<<(N_NODES * HID + 255) / 256, 256, 0, stream>>>(H2, N_NODES * HID);
    spmm_scatter<<<(E + EPB - 1) / EPB, 128, 0, stream>>>(erow, ecol, eval, H1b, H2, E);
    fc2_logsoftmax<<<N_NODES / 16, 128, 0, stream>>>(H2, W2p, b2, Out);
}
